// MiniCPMAttention_78116865180034
// MI455X (gfx1250) — compile-verified, matrix operand order checked by reading, not run on hardware
//
#include <hip/hip_runtime.h>

// ---------------------------------------------------------------------------
// MiniCPM MLA attention forward for gfx1250 (MI455X), wave32 + WMMA f16/f32.
// WMMA operands load as contiguous 16B vectors: weights pre-transposed to
// [N][K], V stored [d][seq] per head.  GEMM templated on tile count so the
// inner loop is branch-free (no spills, clean WMMA register blocks).
// ---------------------------------------------------------------------------

typedef __attribute__((ext_vector_type(16))) _Float16 v16h;
typedef __attribute__((ext_vector_type(8)))  _Float16 v8h;
typedef __attribute__((ext_vector_type(8)))  float    v8f;

#define SEQ      2048
#define HID      2560
#define NH       40
#define QHEAD    96      // 64 nope + 32 rope
#define VHEAD    64
#define QK_ROPE  32
#define QK_NOPE  64

// ---------------- WMMA helpers (wave32, 16x16x32 f16 -> f32) ----------------

__device__ __forceinline__ v8f wmma_f16(v16h a, v16h b, v8f c) {
  return __builtin_amdgcn_wmma_f32_16x16x32_f16(
      /*neg_a=*/false, a, /*neg_b=*/false, b,
      /*c_mod=*/(short)0, c, /*reuse_a=*/false, /*reuse_b=*/false);
}

// Fragment with K contiguous in memory (A of row-major A[M,K], or B of
// transposed Bt[N][K]).  Lanes 0-15: row=lane&15, K k0+0..7 & k0+16..23;
// lanes 16-31: K k0+8..15 & k0+24..31.  ld, k0 multiples of 8 (16B loads).
__device__ __forceinline__ v16h load_frag_k(const _Float16* src, int ld,
                                            int row0, int k0) {
  const int lane = threadIdx.x & 31;
  const int half = lane >> 4;
  const int m    = lane & 15;
  const _Float16* p = src + (size_t)(row0 + m) * ld + k0 + half * 8;
  v8h a0 = *(const v8h*)(p);
  v8h a1 = *(const v8h*)(p + 16);
  v16h a;
#pragma unroll
  for (int i = 0; i < 8; ++i) { a[i] = a0[i]; a[i + 8] = a1[i]; }
  return a;
}

// Same fragment from a per-lane base pointer (k-loop hoisted addressing).
__device__ __forceinline__ v16h load_frag_p(const _Float16* p) {
  v8h a0 = *(const v8h*)(p);
  v8h a1 = *(const v8h*)(p + 16);
  v16h a;
#pragma unroll
  for (int i = 0; i < 8; ++i) { a[i] = a0[i]; a[i + 8] = a1[i]; }
  return a;
}

// ---------------- elementwise f32 -> f16 conversion -------------------------

__global__ void cvt_f32_to_f16(const float* __restrict__ src,
                               _Float16* __restrict__ dst, int n) {
  for (int i = blockIdx.x * blockDim.x + threadIdx.x; i < n;
       i += gridDim.x * blockDim.x)
    dst[i] = (_Float16)src[i];
}

// ---------------- f32 -> f16 convert + transpose ----------------------------
// src f32 [R,C] row-major -> dst f16 [C,R] row-major.  LDS-tiled, 256 threads.

__global__ void cvt_transpose_f16(const float* __restrict__ src,
                                  _Float16* __restrict__ dst, int R, int C) {
  __shared__ _Float16 tile[32][33];
  const int bx = blockIdx.x * 32;              // col base (src)
  const int by = blockIdx.y * 32;              // row base (src)
  const int tx = threadIdx.x & 31;
  const int ty = threadIdx.x >> 5;             // 8 rows per pass
  for (int i = ty; i < 32; i += 8) {
    const int r = by + i, c = bx + tx;
    tile[i][tx] = (r < R && c < C) ? (_Float16)src[(size_t)r * C + c]
                                   : (_Float16)0.f;
  }
  __syncthreads();
  for (int i = ty; i < 32; i += 8) {
    const int c = bx + i, r = by + tx;
    if (c < C && r < R) dst[(size_t)c * R + r] = tile[tx][i];
  }
}

// ---------------- WMMA GEMM: C[M,N] = A[M,K] x Bt[N,K]^T --------------------
// block = 128 threads (4 waves); wave w computes rows m0 = 64*by + 16w and
// NT 16-col tiles starting at n0.  Branch-free inner loop (NT compile-time).
// M multiple of 64, K multiple of 32.

template <int NT>
__global__ __launch_bounds__(128) void gemm_f16_wmma_bt(
    const _Float16* __restrict__ A, const _Float16* __restrict__ Bt,
    float* __restrict__ C, int N, int K, int n0base) {
  const int wave = threadIdx.x >> 5;
  const int m0 = blockIdx.y * 64 + wave * 16;
  const int n0 = n0base + blockIdx.x * (NT * 16);

  const int lane = threadIdx.x & 31;
  const int half = lane >> 4, nl = lane & 15;

  // per-lane base pointers, advanced by 32 each k-step
  const _Float16* pa = A + (size_t)(m0 + nl) * K + half * 8;
  const _Float16* pb[NT];
#pragma unroll
  for (int t = 0; t < NT; ++t)
    pb[t] = Bt + (size_t)(n0 + t * 16 + nl) * K + half * 8;

  v8f acc[NT];
#pragma unroll
  for (int t = 0; t < NT; ++t) acc[t] = (v8f){};

  for (int k = 0; k < K; k += 32) {
    const v16h a = load_frag_p(pa);
    pa += 32;
#pragma unroll
    for (int t = 0; t < NT; ++t) {
      acc[t] = wmma_f16(a, load_frag_p(pb[t]), acc[t]);
      pb[t] += 32;
    }
  }

#pragma unroll
  for (int t = 0; t < NT; ++t)
#pragma unroll
    for (int r = 0; r < 8; ++r)
      C[(size_t)(m0 + r + half * 8) * N + n0 + t * 16 + nl] = acc[t][r];
}

// ---------------- RMS norm (f32 in -> normalized f16 out) -------------------

__global__ void rmsnorm_f16(const float* __restrict__ x,
                            const float* __restrict__ w,
                            _Float16* __restrict__ y,
                            int width, int in_stride, int out_stride) {
  const int row = blockIdx.x;
  __shared__ float red[256];
  float s = 0.f;
  for (int j = threadIdx.x; j < width; j += blockDim.x) {
    float v = x[(size_t)row * in_stride + j];
    s += v * v;
  }
  red[threadIdx.x] = s;
  __syncthreads();
  for (int off = 128; off > 0; off >>= 1) {
    if ((int)threadIdx.x < off) red[threadIdx.x] += red[threadIdx.x + off];
    __syncthreads();
  }
  const float inv = rsqrtf(red[0] / (float)width + 1e-6f);
  for (int j = threadIdx.x; j < width; j += blockDim.x)
    y[(size_t)row * out_stride + j] =
        (_Float16)(x[(size_t)row * in_stride + j] * inv * w[j]);
}

// ---------------- RoPE on k_pe (ckv cols 256..287) -> f16 [S,32] ------------

__global__ void rope_kpe(const float* __restrict__ ckv,
                         const float* __restrict__ cosb,
                         const float* __restrict__ sinb,
                         _Float16* __restrict__ kpe16) {
  const int q = blockIdx.x, j = threadIdx.x;  // block = 32 threads
  const float* kp = ckv + (size_t)q * 288 + 256;
  const float c = cosb[q * QK_ROPE + j], s = sinb[q * QK_ROPE + j];
  const float x = kp[j];
  const float rot = (j < 16) ? -kp[j + 16] : kp[j - 16];
  kpe16[q * QK_ROPE + j] = (_Float16)(x * c + rot * s);
}

// ---------------- assemble per-head Q (RoPE on last 32 dims) ----------------
// qfull f32 [S, NH*96] -> query f16 [NH][S][96]

__global__ void build_query(const float* __restrict__ qfull,
                            const float* __restrict__ cosb,
                            const float* __restrict__ sinb,
                            _Float16* __restrict__ query) {
  const int q = blockIdx.x;
  for (int idx = threadIdx.x; idx < NH * QHEAD; idx += blockDim.x) {
    const int h = idx / QHEAD, d = idx % QHEAD;
    float val = qfull[(size_t)q * (NH * QHEAD) + h * QHEAD + d];
    if (d >= QK_NOPE) {
      const int j = d - QK_NOPE;
      const float c = cosb[q * QK_ROPE + j], s = sinb[q * QK_ROPE + j];
      const int jr = (j < 16) ? j + 16 : j - 16;
      float rot = qfull[(size_t)q * (NH * QHEAD) + h * QHEAD + QK_NOPE + jr];
      if (j < 16) rot = -rot;
      val = val * c + rot * s;
    }
    query[((size_t)h * SEQ + q) * QHEAD + d] = (_Float16)val;
  }
}

// ---------------- assemble per-head K (nope + rope) and V^T -----------------
// kvfull f32 [S, NH*128] -> key f16 [NH][S][96], vT f16 [NH][64][S]

__global__ void build_kv(const float* __restrict__ kvfull,
                         const _Float16* __restrict__ kpe16,
                         _Float16* __restrict__ key,
                         _Float16* __restrict__ vT) {
  const int q = blockIdx.x;
  for (int idx = threadIdx.x; idx < NH * 128; idx += blockDim.x) {
    const int h = idx / 128, d = idx % 128;
    const float val = kvfull[(size_t)q * (NH * 128) + h * 128 + d];
    if (d < QK_NOPE)
      key[((size_t)h * SEQ + q) * QHEAD + d] = (_Float16)val;
    else
      vT[((size_t)h * VHEAD + (d - QK_NOPE)) * SEQ + q] = (_Float16)val;
  }
  for (int idx = threadIdx.x; idx < NH * QK_ROPE; idx += blockDim.x) {
    const int h = idx / QK_ROPE, j = idx % QK_ROPE;
    key[((size_t)h * SEQ + q) * QHEAD + QK_NOPE + j] = kpe16[q * QK_ROPE + j];
  }
}

// ---------------- flash attention, 1 wave per (head, 16-query tile) ---------

__device__ __forceinline__ float half_max16(float v) {
  v = fmaxf(v, __shfl_xor(v, 1, 16));
  v = fmaxf(v, __shfl_xor(v, 2, 16));
  v = fmaxf(v, __shfl_xor(v, 4, 16));
  v = fmaxf(v, __shfl_xor(v, 8, 16));
  return v;
}
__device__ __forceinline__ float half_sum16(float v) {
  v += __shfl_xor(v, 1, 16);
  v += __shfl_xor(v, 2, 16);
  v += __shfl_xor(v, 4, 16);
  v += __shfl_xor(v, 8, 16);
  return v;
}

__global__ __launch_bounds__(32) void mla_flash(
    const _Float16* __restrict__ query, const _Float16* __restrict__ key,
    const _Float16* __restrict__ vTg, _Float16* __restrict__ attn) {
  const int h  = blockIdx.y;
  const int q0 = blockIdx.x * 16;
  const _Float16* Q  = query + (size_t)h * SEQ * QHEAD;
  const _Float16* Kt = key   + (size_t)h * SEQ * QHEAD;
  const _Float16* Vt = vTg   + (size_t)h * VHEAD * SEQ;   // [64][SEQ]

  __shared__ __align__(16) _Float16 pbuf[16 * 32];

  const int lane = threadIdx.x & 31;
  const int half = lane >> 4, nl = lane & 15;

  // Q tile 16x96: three fragments, loaded once.
  const v16h qa0 = load_frag_k(Q, QHEAD, q0, 0);
  const v16h qa1 = load_frag_k(Q, QHEAD, q0, 32);
  const v16h qa2 = load_frag_k(Q, QHEAD, q0, 64);

  v8f o0 = {}, o1 = {}, o2 = {}, o3 = {};
  float rmax[8], rsum[8];
#pragma unroll
  for (int r = 0; r < 8; ++r) { rmax[r] = -1e30f; rsum[r] = 0.f; }

  const float scale = 0.1020620726f;  // 96^-0.5

  for (int kb = 0; kb <= q0 + 15; kb += 32) {
    // scores for 32 keys: two 16x16 tiles (B = K^T, rows K-contiguous)
    v8f s0 = {}, s1 = {};
    s0 = wmma_f16(qa0, load_frag_k(Kt, QHEAD, kb,      0),  s0);
    s0 = wmma_f16(qa1, load_frag_k(Kt, QHEAD, kb,      32), s0);
    s0 = wmma_f16(qa2, load_frag_k(Kt, QHEAD, kb,      64), s0);
    s1 = wmma_f16(qa0, load_frag_k(Kt, QHEAD, kb + 16, 0),  s1);
    s1 = wmma_f16(qa1, load_frag_k(Kt, QHEAD, kb + 16, 32), s1);
    s1 = wmma_f16(qa2, load_frag_k(Kt, QHEAD, kb + 16, 64), s1);

    // scale + causal mask + online softmax update
#pragma unroll
    for (int r = 0; r < 8; ++r) {
      const int qi = q0 + r + half * 8;
      float a = s0[r] * scale;
      float b = s1[r] * scale;
      if (kb + nl > qi)      a = -1e30f;
      if (kb + 16 + nl > qi) b = -1e30f;
      const float tmax = half_max16(fmaxf(a, b));
      const float nm   = fmaxf(rmax[r], tmax);
      const float alpha = __expf(rmax[r] - nm);
      rmax[r] = nm;
      rsum[r] *= alpha;
      o0[r] *= alpha; o1[r] *= alpha; o2[r] *= alpha; o3[r] *= alpha;
      const float p0 = __expf(a - nm);
      const float p1 = __expf(b - nm);
      s0[r] = p0; s1[r] = p1;
      rsum[r] += half_sum16(p0 + p1);
    }

    // re-shape P (C layout) -> A layout via LDS
#pragma unroll
    for (int r = 0; r < 8; ++r) {
      pbuf[(r + half * 8) * 32 + nl]      = (_Float16)s0[r];
      pbuf[(r + half * 8) * 32 + 16 + nl] = (_Float16)s1[r];
    }
    __syncthreads();
    const v16h pa = load_frag_k(pbuf, 32, 0, 0);
    __syncthreads();

    // O += P(16x32) * V(32x64); B fragments contiguous from V^T [64][SEQ]
    o0 = wmma_f16(pa, load_frag_k(Vt, SEQ, 0,  kb), o0);
    o1 = wmma_f16(pa, load_frag_k(Vt, SEQ, 16, kb), o1);
    o2 = wmma_f16(pa, load_frag_k(Vt, SEQ, 32, kb), o2);
    o3 = wmma_f16(pa, load_frag_k(Vt, SEQ, 48, kb), o3);
  }

  // normalize + store to attn [S, NH*VHEAD] (b,q,h,d layout flattened)
#pragma unroll
  for (int r = 0; r < 8; ++r) {
    const int m = r + half * 8;
    const float inv = 1.f / rsum[r];
    _Float16* dst = attn + (size_t)(q0 + m) * (NH * VHEAD) + h * VHEAD;
    dst[nl]      = (_Float16)(o0[r] * inv);
    dst[16 + nl] = (_Float16)(o1[r] * inv);
    dst[32 + nl] = (_Float16)(o2[r] * inv);
    dst[48 + nl] = (_Float16)(o3[r] * inv);
  }
}

// ---------------------------------------------------------------------------

extern "C" void kernel_launch(void* const* d_in, const int* in_sizes, int n_in,
                              void* d_out, int out_size, void* d_ws,
                              size_t ws_size, hipStream_t stream) {
  const float* hs      = (const float*)d_in[0];
  const float* cosb    = (const float*)d_in[1];
  const float* sinb    = (const float*)d_in[2];
  const float* wq_a    = (const float*)d_in[3];
  const float* q_ln_w  = (const float*)d_in[4];
  const float* wq_b    = (const float*)d_in[5];
  const float* wkv_a   = (const float*)d_in[6];
  const float* kv_ln_w = (const float*)d_in[7];
  const float* wkv_b   = (const float*)d_in[8];
  const float* wo      = (const float*)d_in[9];
  float* out = (float*)d_out;

  // workspace carve-up (256B aligned)
  char* wp = (char*)d_ws;
  auto alloc = [&](size_t bytes) -> void* {
    void* p = (void*)wp;
    wp += (bytes + 255) & ~(size_t)255;
    return p;
  };
  _Float16* hs16    = (_Float16*)alloc((size_t)SEQ * HID * 2);
  _Float16* wqaT    = (_Float16*)alloc((size_t)768 * HID * 2);             // [768][2560]
  _Float16* wqbT    = (_Float16*)alloc((size_t)(NH * QHEAD) * 768 * 2);    // [3840][768]
  _Float16* wkvaT   = (_Float16*)alloc((size_t)288 * HID * 2);             // [288][2560]
  _Float16* wkvbT   = (_Float16*)alloc((size_t)(NH * 128) * 256 * 2);      // [5120][256]
  _Float16* woT     = (_Float16*)alloc((size_t)HID * (NH * VHEAD) * 2);    // [2560][2560]
  float*    qa_f32  = (float*)   alloc((size_t)SEQ * 768 * 4);
  float*    ckv_f32 = (float*)   alloc((size_t)SEQ * 288 * 4);
  _Float16* qa16    = (_Float16*)alloc((size_t)SEQ * 768 * 2);
  _Float16* ckv16   = (_Float16*)alloc((size_t)SEQ * 256 * 2);
  _Float16* kpe16   = (_Float16*)alloc((size_t)SEQ * QK_ROPE * 2);
  float*    qfull   = (float*)   alloc((size_t)SEQ * (NH * QHEAD) * 4);
  float*    kvfull  = (float*)   alloc((size_t)SEQ * (NH * 128) * 4);
  _Float16* query16 = (_Float16*)alloc((size_t)NH * SEQ * QHEAD * 2);
  _Float16* key16   = (_Float16*)alloc((size_t)NH * SEQ * QHEAD * 2);
  _Float16* vT16    = (_Float16*)alloc((size_t)NH * VHEAD * SEQ * 2);
  _Float16* attn16  = (_Float16*)alloc((size_t)SEQ * (NH * VHEAD) * 2);

  // hidden states: plain convert (stays row-major, used as A)
  {
    int n = SEQ * HID;
    int blocks = (n + 255) / 256; if (blocks > 4096) blocks = 4096;
    cvt_f32_to_f16<<<blocks, 256, 0, stream>>>(hs, hs16, n);
  }
  // weights: convert + transpose to [N][K]
  auto cvtT = [&](const float* s, _Float16* d, int R, int C) {
    dim3 g((C + 31) / 32, (R + 31) / 32);
    cvt_transpose_f16<<<g, 256, 0, stream>>>(s, d, R, C);
  };
  cvtT(wq_a,  wqaT,  HID, 768);
  cvtT(wq_b,  wqbT,  768, NH * QHEAD);
  cvtT(wkv_a, wkvaT, HID, 288);
  cvtT(wkv_b, wkvbT, 256, NH * 128);
  cvtT(wo,    woT,   NH * VHEAD, HID);

  // GEMM dispatch: full 64-wide column blocks (NT=4) + one ragged launch
  auto gemm = [&](const _Float16* A, const _Float16* Bt, float* C,
                  int M, int N, int K) {
    const int full = N / 64;
    const int remT = (N - full * 64) / 16;     // 0..3 sixteen-col tiles
    if (full > 0)
      gemm_f16_wmma_bt<4><<<dim3(full, M / 64), 128, 0, stream>>>(
          A, Bt, C, N, K, 0);
    if (remT == 1)
      gemm_f16_wmma_bt<1><<<dim3(1, M / 64), 128, 0, stream>>>(
          A, Bt, C, N, K, full * 64);
    else if (remT == 2)
      gemm_f16_wmma_bt<2><<<dim3(1, M / 64), 128, 0, stream>>>(
          A, Bt, C, N, K, full * 64);
    else if (remT == 3)
      gemm_f16_wmma_bt<3><<<dim3(1, M / 64), 128, 0, stream>>>(
          A, Bt, C, N, K, full * 64);
  };

  // projections from hidden states
  gemm(hs16, wqaT,  qa_f32,  SEQ, 768, HID);
  gemm(hs16, wkvaT, ckv_f32, SEQ, 288, HID);

  // RMS norms
  rmsnorm_f16<<<SEQ, 256, 0, stream>>>(qa_f32, q_ln_w, qa16, 768, 768, 768);
  rmsnorm_f16<<<SEQ, 256, 0, stream>>>(ckv_f32, kv_ln_w, ckv16, 256, 288, 256);

  // RoPE on k_pe
  rope_kpe<<<SEQ, 32, 0, stream>>>(ckv_f32, cosb, sinb, kpe16);

  // second-stage projections
  gemm(qa16,  wqbT,  qfull,  SEQ, NH * QHEAD, 768);
  gemm(ckv16, wkvbT, kvfull, SEQ, NH * 128,   256);

  // per-head Q/K/V assembly (RoPE on q_pe, k_pe broadcast, V transposed)
  build_query<<<SEQ, 256, 0, stream>>>(qfull, cosb, sinb, query16);
  build_kv<<<SEQ, 256, 0, stream>>>(kvfull, kpe16, key16, vT16);

  // flash attention: grid (128 query tiles, 40 heads), 1 wave per block
  mla_flash<<<dim3(SEQ / 16, NH), 32, 0, stream>>>(query16, key16, vT16,
                                                   attn16);

  // output projection -> d_out (f32)
  gemm(attn16, woT, out, SEQ, HID, NH * VHEAD);
}
